// TransformerEncoder_32736240730231
// MI455X (gfx1250) — matrix operand order repaired and checked by reading, not compiled, not run on hardware
//
#include <hip/hip_runtime.h>
#include <math.h>

typedef __attribute__((ext_vector_type(16))) _Float16 v16h;
typedef __attribute__((ext_vector_type(8)))  float    v8f;

namespace te {

constexpr int BATCH = 8, SEQ = 1024, F = 768, NH = 12, DH = 64, MHID = 3072;
constexpr int MROWS = BATCH * SEQ;            // 8192 token rows
constexpr float SM_SCALE = 0.03608439182435161f; // 1/sqrt(768)

union HV16 { v16h h; uint4 u[2]; };

__device__ __forceinline__ v8f wmma16(v16h a, v16h b, v8f c) {
  // v_wmma_f32_16x16x32_f16: D = A(16x32) * B(32x16) + C(16x16 f32)
  return __builtin_amdgcn_wmma_f32_16x16x32_f16(false, a, false, b, (short)0, c,
                                                false, false);
}

__device__ __forceinline__ v8f vzero8() {
  v8f z;
#pragma unroll
  for (int i = 0; i < 8; ++i) z[i] = 0.f;
  return z;
}

// ---- DPP16 butterfly reduction within each 16-lane half (no LDS traffic) ----
// 0xB1 = quad_perm(1,0,3,2) (xor1), 0x4E = quad_perm(2,3,0,1) (xor2),
// 0x141 = row_half_mirror, 0x140 = row_mirror. After each stage groups are
// uniform, so the mirrors complete the xor4/xor8 combines within 16 lanes.
template <int CTRL>
__device__ __forceinline__ float dpp_f(float x) {
  const int xi = __float_as_int(x);
  return __int_as_float(__builtin_amdgcn_update_dpp(xi, xi, CTRL, 0xF, 0xF, true));
}
__device__ __forceinline__ float red_max16(float x) {
  x = fmaxf(x, dpp_f<0xB1>(x));
  x = fmaxf(x, dpp_f<0x4E>(x));
  x = fmaxf(x, dpp_f<0x141>(x));
  x = fmaxf(x, dpp_f<0x140>(x));
  return x;
}
__device__ __forceinline__ float red_sum16(float x) {
  x += dpp_f<0xB1>(x);
  x += dpp_f<0x4E>(x);
  x += dpp_f<0x141>(x);
  x += dpp_f<0x140>(x);
  return x;
}

// A-fragment (16x32 f16): lane m = lane%16, K-halves [base..base+15]
// contiguous, base = 16*(lane>=16). Two 16B chunks -> 2 x b128.
__device__ __forceinline__ v16h frag_a(const _Float16* row16, int ld, int lane) {
  const int m = lane & 15, base = (lane >> 4) * 16;
  const _Float16* p = row16 + m * ld + base;
  HV16 a;
  a.u[0] = *(const uint4*)(p);
  a.u[1] = *(const uint4*)(p + 8);
  return a.h;
}

// B-fragment where the lane's 16 K-values are CONTIGUOUS at p (K-major).
__device__ __forceinline__ v16h frag_kmajor(const _Float16* p) {
  HV16 b;
  b.u[0] = *(const uint4*)(p);
  b.u[1] = *(const uint4*)(p + 8);
  return b.h;
}

// B-fragment from K-major LDS tile: col n = lane%16, K base = 16*(lane>=16).
__device__ __forceinline__ v16h frag_bT_lds(const _Float16* s, int ldt, int lane) {
  const int n = lane & 15, kb = (lane >> 4) * 16;
  return frag_kmajor(s + n * ldt + kb);
}

__device__ __forceinline__ float gelu_exact(float x) {
  return 0.5f * x * (1.0f + erff(x * 0.70710678118654752f));
}

// -------------------------------------------------- weight convert+transpose
// in: f32 [K x N] row-major  ->  out: f16 [N x K] (K-major for WMMA B-frags)
__global__ void cvt_f16_t(const float* __restrict__ in, _Float16* __restrict__ out,
                          int K, int N) {
  int i = blockIdx.x * blockDim.x + threadIdx.x;
  if (i < K * N) {
    const int kIdx = i / N, nIdx = i - kIdx * N;
    out[(size_t)nIdx * K + kIdx] = (_Float16)in[i];
  }
}

// ---------------------------------------------------------------- layernorm
__global__ __launch_bounds__(256)
void layernorm768(const float* __restrict__ x, const float* __restrict__ g,
                  const float* __restrict__ b, _Float16* __restrict__ out) {
  const int row = blockIdx.x, t = threadIdx.x;
  const int lane = t & 31, wv = t >> 5;
  const float* xr = x + (size_t)row * F;
  float v0 = xr[t], v1 = xr[t + 256], v2 = xr[t + 512];

  __shared__ float red[8];
  __shared__ float stat[2];

  float s = v0 + v1 + v2;
#pragma unroll
  for (int m = 16; m >= 1; m >>= 1) s += __shfl_xor(s, m, 32);
  if (lane == 0) red[wv] = s;
  __syncthreads();
  if (t == 0) {
    float a = 0.f;
    for (int i = 0; i < 8; ++i) a += red[i];
    stat[0] = a * (1.0f / 768.0f);
  }
  __syncthreads();
  const float mean = stat[0];
  float d0 = v0 - mean, d1 = v1 - mean, d2 = v2 - mean;
  float q = d0 * d0 + d1 * d1 + d2 * d2;
#pragma unroll
  for (int m = 16; m >= 1; m >>= 1) q += __shfl_xor(q, m, 32);
  if (lane == 0) red[wv] = q;
  __syncthreads();
  if (t == 0) {
    float a = 0.f;
    for (int i = 0; i < 8; ++i) a += red[i];
    stat[1] = rsqrtf(a * (1.0f / 768.0f) + 1e-5f);
  }
  __syncthreads();
  const float rstd = stat[1];
  _Float16* o = out + (size_t)row * F;
  o[t]       = (_Float16)(d0 * rstd * g[t]       + b[t]);
  o[t + 256] = (_Float16)(d1 * rstd * g[t + 256] + b[t + 256]);
  o[t + 512] = (_Float16)(d2 * rstd * g[t + 512] + b[t + 512]);
}

// ---------------------------------------------------------------- GEMM
// C[M,N] = A[M,K] * WT[N,K]^T (+bias, fused epilogue). M = 8192 always.
// 256 threads = 8 waves; 128x128 tile, K-step 32, register-staged LDS
// double buffer (one barrier per K-step). Wave (wm,wn) owns 32x64 = 2x4
// WMMA tiles. LDS tiles are K-major, ld=40 halves (80B stride: bank-clean).
enum { GM_QKV = 0, GM_QKVT = 1, GM_WO = 2, GM_FFN1 = 3, GM_FFN2 = 4 };

template <int MODE>
__global__ __launch_bounds__(256)
void gemm128(const _Float16* __restrict__ A, const _Float16* __restrict__ WT,
             const float* __restrict__ bias, const float* __restrict__ resid,
             float* __restrict__ outF, _Float16* __restrict__ outH,
             int Nsz, int Ksz) {
  constexpr int BM = 128, BN = 128, BK = 32, LD = BK + 8;
  constexpr int TSZ = BM * LD;  // halves per buffer
  __shared__ __align__(16) _Float16 As[2 * TSZ];   // [buf][row][k]
  __shared__ __align__(16) _Float16 BsT[2 * TSZ];  // [buf][col][k]

  const int t = threadIdx.x, lane = t & 31, wave = t >> 5;
  const int wm = wave >> 1, wn = wave & 1;
  const int tileM = blockIdx.y * BM, tileN = blockIdx.x * BN;

  const int lrow = t >> 1, lkb = (t & 1) * 16;  // 128 rows/cols x 32 K-halves
  const _Float16* gA = A  + (size_t)(tileM + lrow) * Ksz + lkb;
  const _Float16* gB = WT + (size_t)(tileN + lrow) * Ksz + lkb;
  _Float16* sA = As  + lrow * LD + lkb;
  _Float16* sB = BsT + lrow * LD + lkb;

  v8f acc[2][4];
#pragma unroll
  for (int i = 0; i < 2; ++i)
#pragma unroll
    for (int j = 0; j < 4; ++j) acc[i][j] = vzero8();

  const int nk = Ksz / BK;

  // prologue: stage tile 0
  uint4 ra0 = *(const uint4*)(gA);
  uint4 ra1 = *(const uint4*)(gA + 8);
  uint4 rb0 = *(const uint4*)(gB);
  uint4 rb1 = *(const uint4*)(gB + 8);
  *(uint4*)(sA)     = ra0; *(uint4*)(sA + 8) = ra1;
  *(uint4*)(sB)     = rb0; *(uint4*)(sB + 8) = rb1;
  __syncthreads();

  for (int kt = 0; kt < nk; ++kt) {
    const int cur = kt & 1;
    if (kt + 1 < nk) {  // global loads for next tile into registers
      const int k1 = (kt + 1) * BK;
      ra0 = *(const uint4*)(gA + k1);
      ra1 = *(const uint4*)(gA + k1 + 8);
      rb0 = *(const uint4*)(gB + k1);
      rb1 = *(const uint4*)(gB + k1 + 8);
      if (kt + 2 < nk) {  // distance-2 speculative prefetch
        __builtin_prefetch(gA + k1 + BK, 0, 1);
        __builtin_prefetch(gB + k1 + BK, 0, 1);
      }
    }

    const _Float16* Ab = As  + cur * TSZ;
    const _Float16* Bb = BsT + cur * TSZ;
    const v16h af0 = frag_a(Ab + (wm * 32) * LD, LD, lane);
    const v16h af1 = frag_a(Ab + (wm * 32 + 16) * LD, LD, lane);
    v16h bf[4];
#pragma unroll
    for (int ni = 0; ni < 4; ++ni)
      bf[ni] = frag_bT_lds(Bb + (wn * 64 + ni * 16) * LD, LD, lane);
#pragma unroll
    for (int ni = 0; ni < 4; ++ni) {
      acc[0][ni] = wmma16(af0, bf[ni], acc[0][ni]);
      acc[1][ni] = wmma16(af1, bf[ni], acc[1][ni]);
    }

    if (kt + 1 < nk) {  // commit staged regs to the other buffer
      const int nb = cur ^ 1;
      *(uint4*)(sA + nb * TSZ)     = ra0;
      *(uint4*)(sA + nb * TSZ + 8) = ra1;
      *(uint4*)(sB + nb * TSZ)     = rb0;
      *(uint4*)(sB + nb * TSZ + 8) = rb1;
    }
    __syncthreads();
  }

  // Epilogue. C layout: reg r -> row r + 8*(lane>=16); col = lane%16.
  const int hi = lane >> 4, nl = lane & 15;
#pragma unroll
  for (int mi = 0; mi < 2; ++mi)
#pragma unroll
    for (int ni = 0; ni < 4; ++ni)
#pragma unroll
      for (int r = 0; r < 8; ++r) {
        const int rowg = tileM + wm * 32 + mi * 16 + hi * 8 + r;
        const int colg = tileN + wn * 64 + ni * 16 + nl;
        float val = acc[mi][ni][r];
        if (MODE == GM_QKV) {          // q/k -> [b,h,n,d]
          val += bias[colg];
          const int bb = rowg >> 10, nn = rowg & 1023;
          const int hh = colg >> 6, dd = colg & 63;
          outH[(((size_t)bb * NH + hh) * SEQ + nn) * DH + dd] = (_Float16)val;
        } else if (MODE == GM_QKVT) {  // v -> [b,h,d,n] (K-major for P.V)
          val += bias[colg];
          const int bb = rowg >> 10, nn = rowg & 1023;
          const int hh = colg >> 6, dd = colg & 63;
          outH[(((size_t)bb * NH + hh) * DH + dd) * SEQ + nn] = (_Float16)val;
        } else if (MODE == GM_WO) {
          const size_t idx = (size_t)rowg * F + colg;
          outF[idx] = val + bias[colg] + resid[idx];
        } else if (MODE == GM_FFN1) {
          outH[(size_t)rowg * MHID + colg] = (_Float16)gelu_exact(val + bias[colg]);
        } else {  // GM_FFN2 -> final output
          const size_t idx = (size_t)rowg * F + colg;
          outF[idx] = gelu_exact(val + bias[colg]) + resid[idx];
        }
      }
}

// ---------------------------------------------------------------- attention
// Flash-style attention. Block = 4 waves; wave owns 16 query rows of one
// (b,h); streams 32-key blocks. grid = (SEQ/64, B*H).
// q,k: [b,h,n,64];  vT: [b,h,64,n] so P.V B-fragments are K(=key)-major.
// Softmax scale folded into Q fragments; reductions use DPP16 butterflies.
__global__ __launch_bounds__(128)
void flash_attn(const _Float16* __restrict__ q, const _Float16* __restrict__ k,
                const _Float16* __restrict__ vT, _Float16* __restrict__ attn) {
  const int bh = blockIdx.y;
  const int wave = threadIdx.x >> 5, lane = threadIdx.x & 31;
  const int hi = lane >> 4, nlane = lane & 15, mrow = lane & 15, abase = hi * 16;
  const int qr0 = blockIdx.x * 64 + wave * 16;
  const int bb = bh / NH, hh = bh % NH;
  const _Float16* qb  = q  + (size_t)bh * SEQ * DH;
  const _Float16* kb  = k  + (size_t)bh * SEQ * DH;
  const _Float16* vTb = vT + (size_t)bh * DH * SEQ;

  __shared__ __align__(16) _Float16 Pl[4][16][40];  // per-wave P staging

  // Q A-fragments with softmax scale pre-applied (f16 packed muls, once)
  v16h qa0, qa1;
  {
    const _Float16* p = qb + (size_t)(qr0 + mrow) * DH;
    HV16 a0, a1;
    a0.u[0] = *(const uint4*)(p + abase);
    a0.u[1] = *(const uint4*)(p + abase + 8);
    a1.u[0] = *(const uint4*)(p + 32 + abase);
    a1.u[1] = *(const uint4*)(p + 32 + abase + 8);
    qa0 = a0.h; qa1 = a1.h;
    const _Float16 hs = (_Float16)SM_SCALE;
#pragma unroll
    for (int i = 0; i < 16; ++i) { qa0[i] *= hs; qa1[i] *= hs; }
  }

  float mi[8], li[8];
  v8f o[4];
#pragma unroll
  for (int r = 0; r < 8; ++r) { mi[r] = -1e30f; li[r] = 0.f; }
#pragma unroll
  for (int f = 0; f < 4; ++f) o[f] = vzero8();

  for (int j0 = 0; j0 < SEQ; j0 += 32) {
    // ---- S(16x32) = (Q*scale) Kt : B-frag col = key, K-dim = d ----
    v8f s0 = vzero8(), s1 = vzero8();
    {
      const _Float16* p0 = kb + (size_t)(j0 + nlane) * DH + hi * 16;
      const _Float16* p1 = kb + (size_t)(j0 + 16 + nlane) * DH + hi * 16;
      s0 = wmma16(qa0, frag_kmajor(p0), s0);
      s0 = wmma16(qa1, frag_kmajor(p0 + 32), s0);
      s1 = wmma16(qa0, frag_kmajor(p1), s1);
      s1 = wmma16(qa1, frag_kmajor(p1 + 32), s1);
    }

    // ---- online softmax (rows in 16-lane halves; DPP16 butterflies) ----
#pragma unroll
    for (int r = 0; r < 8; ++r) {
      const float tmax = red_max16(fmaxf(s0[r], s1[r]));
      const float nm = fmaxf(mi[r], tmax);
      const float corr = __expf(mi[r] - nm);
      li[r] *= corr;
#pragma unroll
      for (int f = 0; f < 4; ++f) o[f][r] *= corr;
      const float p0e = __expf(s0[r] - nm);
      const float p1e = __expf(s1[r] - nm);
      s0[r] = p0e; s1[r] = p1e;
      li[r] += red_sum16(p0e + p1e);
      mi[r] = nm;
    }

    // ---- C-layout -> A-layout via per-wave LDS tile ----
#pragma unroll
    for (int r = 0; r < 8; ++r) {
      Pl[wave][hi * 8 + r][nlane]      = (_Float16)s0[r];
      Pl[wave][hi * 8 + r][16 + nlane] = (_Float16)s1[r];
    }
    v16h pa;  // LDS ops are in-order within a wave
    {
      const _Float16* p = &Pl[wave][mrow][0];
      HV16 a;
      a.u[0] = *(const uint4*)(p + abase);
      a.u[1] = *(const uint4*)(p + abase + 8);
      pa = a.h;
    }

    // ---- O(16x64) += P(16x32) V(32x64); vT rows are key-contiguous ----
#pragma unroll
    for (int f = 0; f < 4; ++f) {
      const _Float16* p = vTb + (size_t)(f * 16 + nlane) * SEQ + j0 + hi * 16;
      o[f] = wmma16(pa, frag_kmajor(p), o[f]);
    }
  }

  // normalize and emit in [b, n, h*64+d] (ready for Wo GEMM)
#pragma unroll
  for (int f = 0; f < 4; ++f)
#pragma unroll
    for (int r = 0; r < 8; ++r) {
      const int rowg = qr0 + hi * 8 + r;
      const float val = o[f][r] / li[r];
      attn[((size_t)(bb * SEQ + rowg)) * F + hh * DH + f * 16 + nlane] =
          (_Float16)val;
    }
}

}  // namespace te

extern "C" void kernel_launch(void* const* d_in, const int* in_sizes, int n_in,
                              void* d_out, int out_size, void* d_ws, size_t ws_size,
                              hipStream_t stream) {
  using namespace te;
  (void)in_sizes; (void)n_in; (void)out_size; (void)ws_size;

  const float* x     = (const float*)d_in[0];
  const float* Wq    = (const float*)d_in[1];
  const float* bq    = (const float*)d_in[2];
  const float* Wk    = (const float*)d_in[3];
  const float* bk    = (const float*)d_in[4];
  const float* Wv    = (const float*)d_in[5];
  const float* bv    = (const float*)d_in[6];
  const float* Wo    = (const float*)d_in[7];
  const float* bo    = (const float*)d_in[8];
  const float* ln1_g = (const float*)d_in[9];
  const float* ln1_b = (const float*)d_in[10];
  const float* ln2_g = (const float*)d_in[11];
  const float* ln2_b = (const float*)d_in[12];
  const float* W1    = (const float*)d_in[13];
  const float* b1    = (const float*)d_in[14];
  const float* W2    = (const float*)d_in[15];
  const float* b2    = (const float*)d_in[16];

  // ---- workspace carve-up (256B aligned) ----
  char* wsp = (char*)d_ws;
  size_t off = 0;
  auto take = [&](size_t bytes) -> char* {
    off = (off + 255) & ~(size_t)255;
    char* p = wsp + off;
    off += bytes;
    return p;
  };
  _Float16* WqT  = (_Float16*)take((size_t)F * F * 2);      // [N][K] f16
  _Float16* WkT  = (_Float16*)take((size_t)F * F * 2);
  _Float16* WvT  = (_Float16*)take((size_t)F * F * 2);
  _Float16* WoT  = (_Float16*)take((size_t)F * F * 2);
  _Float16* W1T  = (_Float16*)take((size_t)F * MHID * 2);
  _Float16* W2T  = (_Float16*)take((size_t)MHID * F * 2);
  _Float16* xn16 = (_Float16*)take((size_t)MROWS * F * 2);
  _Float16* q16  = (_Float16*)take((size_t)MROWS * F * 2);  // [b,h,n,d]
  _Float16* k16  = (_Float16*)take((size_t)MROWS * F * 2);  // [b,h,n,d]
  _Float16* vT16 = (_Float16*)take((size_t)MROWS * F * 2);  // [b,h,d,n]
  _Float16* at16 = (_Float16*)take((size_t)MROWS * F * 2);
  float*    res32= (float*)   take((size_t)MROWS * F * 4);
  _Float16* h16  = (_Float16*)take((size_t)MROWS * F * 2);
  _Float16* m16  = (_Float16*)take((size_t)MROWS * MHID * 2);

  // ---- weight convert + transpose to K-major f16 ----
  const int nFF = F * F, nFM = F * MHID;
  cvt_f16_t<<<(nFF + 255) / 256, 256, 0, stream>>>(Wq, WqT, F, F);
  cvt_f16_t<<<(nFF + 255) / 256, 256, 0, stream>>>(Wk, WkT, F, F);
  cvt_f16_t<<<(nFF + 255) / 256, 256, 0, stream>>>(Wv, WvT, F, F);
  cvt_f16_t<<<(nFF + 255) / 256, 256, 0, stream>>>(Wo, WoT, F, F);
  cvt_f16_t<<<(nFM + 255) / 256, 256, 0, stream>>>(W1, W1T, F, MHID);
  cvt_f16_t<<<(nFM + 255) / 256, 256, 0, stream>>>(W2, W2T, MHID, F);

  const dim3 blk(256);
  const dim3 gF(F / 128, MROWS / 128);       // (6, 64)
  const dim3 gM(MHID / 128, MROWS / 128);    // (24, 64)

  // 1) LN1
  layernorm768<<<MROWS, blk, 0, stream>>>(x, ln1_g, ln1_b, xn16);
  // 2) QKV projections (head-split scatter in epilogue; V transposed)
  gemm128<GM_QKV><<<gF, blk, 0, stream>>>(xn16, WqT, bq, nullptr, nullptr, q16, F, F);
  gemm128<GM_QKV><<<gF, blk, 0, stream>>>(xn16, WkT, bk, nullptr, nullptr, k16, F, F);
  gemm128<GM_QKVT><<<gF, blk, 0, stream>>>(xn16, WvT, bv, nullptr, nullptr, vT16, F, F);
  // 3) attention
  flash_attn<<<dim3(SEQ / 64, BATCH * NH), dim3(128), 0, stream>>>(q16, k16, vT16, at16);
  // 4) output projection + residual (fp32 stream)
  gemm128<GM_WO><<<gF, blk, 0, stream>>>(at16, WoT, bo, x, res32, nullptr, F, F);
  // 5) LN2
  layernorm768<<<MROWS, blk, 0, stream>>>(res32, ln2_g, ln2_b, h16);
  // 6) FFN1 + GELU
  gemm128<GM_FFN1><<<gM, blk, 0, stream>>>(h16, W1T, b1, nullptr, nullptr, m16, MHID, F);
  // 7) FFN2 + GELU + residual -> final output
  gemm128<GM_FFN2><<<gF, blk, 0, stream>>>(m16, W2T, b2, res32, (float*)d_out, nullptr, F, MHID);
}
